// KalmanNetNN_44719199486689
// MI455X (gfx1250) — matrix operand order repaired and checked by reading, not compiled, not run on hardware
//
#include <hip/hip_runtime.h>
#include <hip/hip_bf16.h>
#include <math.h>

typedef __attribute__((ext_vector_type(16))) _Float16 v16h;
typedef __attribute__((ext_vector_type(8)))  float    v8f;

// ---- problem dims ----
#define MM    4
#define NNY   48
#define TT    256
#define H1S   4160
#define HIDS  2320
#define H2S   768
#define OUTR  192           // rows of W3 (= M*N)
#define HPAD  2336          // HID padded to 73*32
#define JBLK  145           // HID/16

// ---- tile grids (rows/16 x ceil(cols/32)) ----
#define RB1 260
#define KB1 2
#define RBI 435
#define KBI 130
#define RBH 435
#define KBH 73
#define RB2 48
#define KB2 73
#define RB3 12
#define KB3 24

#define GWG   128           // persistent workgroups
#define NWAVE 8             // waves per 256-thread WG (wave32)

__device__ __forceinline__ v8f wmma16(v16h a, v16h b, v8f c) {
  // D = A(16x32 f16) * B(32x16 f16) + C(16x16 f32)
  return __builtin_amdgcn_wmma_f32_16x16x32_f16(false, a, false, b, (short)0, c, false, false);
}
__device__ __forceinline__ v16h ldfrag(const _Float16* p) { return *(const v16h*)p; }

// ---------------- weight prep: fp32 row-major -> fp16 WMMA-A tiles ----------------
// Tile (rb,kb): 32 lanes x 16 f16, lane-contiguous 32B. 16-bit A 16x32, contiguous k order (B loads match):
// lane = half*16+m; element e (pair p=e>>1, sub=e&1): k_local = half*16 + p*2 + sub.
__global__ void __launch_bounds__(256) knet_prep(const float* __restrict__ W,
                                                 int rows, int cols, int RB, int KB,
                                                 _Float16* __restrict__ out) {
  long i = (long)blockIdx.x * blockDim.x + threadIdx.x;
  long ntile = (long)RB * KB;
  if (i >= ntile * 32) return;
  int  lane = (int)(i & 31);
  long tile = i >> 5;
  int  kb = (int)(tile % KB);
  int  rb = (int)(tile / KB);
  int  half = lane >> 4, m = lane & 15;
  int  row = rb * 16 + m;
  _Float16 vals[16] __attribute__((aligned(32)));
#pragma unroll
  for (int e = 0; e < 16; ++e) {
    int p = e >> 1, sub = e & 1;
    int k_local = (half << 4) + (p << 1) + sub;
    int k = kb * 32 + k_local;
    float f = (row < rows && k < cols) ? W[(long)row * cols + k] : 0.0f;
    vals[e] = (_Float16)f;
  }
  *(v16h*)(out + tile * 512 + (long)lane * 16) = *(const v16h*)vals;
}

// ---------------- init: h0 -> buffers, zero pads + barrier ctrl ----------------
__global__ void __launch_bounds__(256) knet_init(const float* __restrict__ h0,
                                                 float* __restrict__ h32a,
                                                 _Float16* __restrict__ h16a,
                                                 _Float16* __restrict__ h16b,
                                                 unsigned* __restrict__ ctrl) {
  int i = blockIdx.x * blockDim.x + threadIdx.x;
  if (i < HIDS) { float v = h0[i]; h32a[i] = v; h16a[i] = (_Float16)v; }
  if (i >= HIDS && i < HPAD) { h16a[i] = (_Float16)0.0f; h16b[i] = (_Float16)0.0f; }
  if (i == 0) { ctrl[0] = 0u; ctrl[1] = 0u; }
}

// ---------------- grid-wide barrier (agent-scope atomics + s_sleep backoff) ----------------
__device__ __forceinline__ void gbar(unsigned* cnt, unsigned* rel, unsigned ep) {
  __threadfence();
  __syncthreads();
  if (threadIdx.x == 0) {
    unsigned prev = __hip_atomic_fetch_add(cnt, 1u, __ATOMIC_ACQ_REL, __HIP_MEMORY_SCOPE_AGENT);
    if (prev == (unsigned)(GWG - 1)) {
      __hip_atomic_store(cnt, 0u, __ATOMIC_RELEASE, __HIP_MEMORY_SCOPE_AGENT);
      __hip_atomic_fetch_add(rel, 1u, __ATOMIC_ACQ_REL, __HIP_MEMORY_SCOPE_AGENT);
    } else {
      while (__hip_atomic_load(rel, __ATOMIC_ACQUIRE, __HIP_MEMORY_SCOPE_AGENT) < ep)
        __builtin_amdgcn_s_sleep(2);
    }
  }
  __syncthreads();
}

// ---------------- scalar head (WG0 only): prior, residual, normalized features ----------------
__device__ void phaseA(int tt,
                       const float* __restrict__ yseq, const float* __restrict__ Aa,
                       const float* __restrict__ Cc,
                       float* s_xpost, float* s_xprior, float* s_resid, float* s_d4, float* s_inv,
                       _Float16* __restrict__ kgin) {
  int tid = threadIdx.x;
  if (tid < MM) {
    float s = 0.f;
#pragma unroll
    for (int k = 0; k < MM; ++k) s += Aa[tid * MM + k] * s_xpost[k];
    s_xprior[tid] = s;
  }
  __syncthreads();
  if (tid < NNY) {
    float s = Cc[tid * (MM + 1) + MM];
#pragma unroll
    for (int k = 0; k < MM; ++k) s += Cc[tid * (MM + 1) + k] * s_xprior[k];
    s_resid[tid] = yseq[tid * TT + tt] - s;
  }
  __syncthreads();
  if (tid == 0) {
    float s2 = 0.f;
    for (int n = 0; n < NNY; ++n) s2 += s_resid[n] * s_resid[n];
    float s4 = 0.f;
    for (int m = 0; m < MM; ++m) { float d = s_xpost[m] - s_xprior[m]; s_d4[m] = d; s4 += d * d; }
    s_inv[0] = 1.f / fmaxf(sqrtf(s2), 1e-12f);
    s_inv[1] = 1.f / fmaxf(sqrtf(s4), 1e-12f);
  }
  __syncthreads();
  if (tid < NNY)                kgin[tid] = (_Float16)(s_resid[tid] * s_inv[0]);
  else if (tid < NNY + MM)      kgin[tid] = (_Float16)(s_d4[tid - NNY] * s_inv[1]);
  else if (tid < 64)            kgin[tid] = (_Float16)0.0f;
  __syncthreads();
}

// ---------------- persistent cooperative kernel ----------------
__global__ void __launch_bounds__(256) knet_main(
    const float* __restrict__ yseq, const float* __restrict__ x0,
    const float* __restrict__ Aa, const float* __restrict__ Cc,
    const float* __restrict__ b1, const float* __restrict__ b_ih,
    const float* __restrict__ b_hh, const float* __restrict__ b2,
    const float* __restrict__ b3,
    unsigned* __restrict__ ctrl,
    _Float16* __restrict__ kgin, _Float16* __restrict__ l1a, _Float16* __restrict__ l2a,
    _Float16* __restrict__ h16x, _Float16* __restrict__ h16y,
    float* __restrict__ h32x, float* __restrict__ h32y,
    const _Float16* __restrict__ w1t, const _Float16* __restrict__ wiht,
    const _Float16* __restrict__ whht, const _Float16* __restrict__ w2t,
    const _Float16* __restrict__ w3t,
    float* __restrict__ dout) {
  __shared__ float s_part[6][NWAVE][16];
  __shared__ float s_kg[OUTR];
  __shared__ float s_xpost[MM], s_xprior[MM], s_resid[NNY], s_d4[MM], s_inv[2];

  unsigned* cnt = ctrl + 0;
  unsigned* rel = ctrl + 1;
  const int tid  = threadIdx.x;
  const int lane = tid & 31;
  const int half = lane >> 4;
  const int wav  = tid >> 5;
  const size_t loff = (size_t)lane * 16;
  const size_t wstride = (size_t)NWAVE * 512;
  unsigned ep = 0;

  if (blockIdx.x == 0) {
    if (tid < MM) s_xpost[tid] = x0[tid];
    __syncthreads();
    phaseA(0, yseq, Aa, Cc, s_xpost, s_xprior, s_resid, s_d4, s_inv, kgin);
  }
  gbar(cnt, rel, ++ep);

  for (int t = 0; t < TT; ++t) {
    const int par = t & 1;
    const _Float16* hcur   = par ? h16y : h16x;
    _Float16*       hnxt   = par ? h16x : h16y;
    const float*    hcur32 = par ? h32y : h32x;
    float*          hnxt32 = par ? h32x : h32y;

    // ---- Phase B: l1 = relu(W1 @ kg_in + b1) ----
    {
      int gw = blockIdx.x * NWAVE + wav;
      for (int rb = gw; rb < RB1; rb += GWG * NWAVE) {
        v16h a0 = ldfrag(w1t + (size_t)(rb * KB1 + 0) * 512 + loff);
        v16h a1 = ldfrag(w1t + (size_t)(rb * KB1 + 1) * 512 + loff);
        v16h b0 = ldfrag(kgin + half * 16);
        v16h b1f = ldfrag(kgin + 32 + half * 16);
        v8f acc = {};
        acc = wmma16(a0, b0, acc);
        acc = wmma16(a1, b1f, acc);
        if ((lane & 15) == 0) {
#pragma unroll
          for (int r = 0; r < 8; ++r) {
            int row = rb * 16 + half * 8 + r;
            float v = acc[r] + b1[row];
            l1a[row] = (_Float16)(v > 0.f ? v : 0.f);
          }
        }
      }
    }
    gbar(cnt, rel, ++ep);

    // ---- Phase C: GRU step; j-blocks over WGs; fused triple-WMMA, sw-pipelined ----
    for (int jb = blockIdx.x; jb < JBLK; jb += GWG) {
      // group 0: ir/iz/ig = W_ih(rows {0,1,2}*HID + jb*16 ..) @ l1
      {
        v8f ar = {}, az = {}, ag = {};
        int kb = wav;
        if (kb < KBI) {
          const _Float16* p0 = wiht + (((size_t)(0 * JBLK + jb) * KBI + kb) * 512) + loff;
          const _Float16* p1 = wiht + (((size_t)(1 * JBLK + jb) * KBI + kb) * 512) + loff;
          const _Float16* p2 = wiht + (((size_t)(2 * JBLK + jb) * KBI + kb) * 512) + loff;
          v16h a0 = ldfrag(p0), a1 = ldfrag(p1), a2 = ldfrag(p2);
          v16h b = ldfrag(l1a + kb * 32 + half * 16);
          for (kb += NWAVE; kb < KBI; kb += NWAVE) {
            p0 += wstride; p1 += wstride; p2 += wstride;
            v16h n0 = ldfrag(p0), n1 = ldfrag(p1), n2 = ldfrag(p2);
            v16h nb = ldfrag(l1a + kb * 32 + half * 16);
            ar = wmma16(a0, b, ar); az = wmma16(a1, b, az); ag = wmma16(a2, b, ag);
            a0 = n0; a1 = n1; a2 = n2; b = nb;
          }
          ar = wmma16(a0, b, ar); az = wmma16(a1, b, az); ag = wmma16(a2, b, ag);
        }
        if ((lane & 15) == 0) {
#pragma unroll
          for (int r = 0; r < 8; ++r) {
            s_part[0][wav][half * 8 + r] = ar[r];
            s_part[1][wav][half * 8 + r] = az[r];
            s_part[2][wav][half * 8 + r] = ag[r];
          }
        }
      }
      // group 1: hr/hz/hg = W_hh(rows {0,1,2}*HID + jb*16 ..) @ h
      {
        v8f ar = {}, az = {}, ag = {};
        int kb = wav;
        if (kb < KBH) {
          const _Float16* p0 = whht + (((size_t)(0 * JBLK + jb) * KBH + kb) * 512) + loff;
          const _Float16* p1 = whht + (((size_t)(1 * JBLK + jb) * KBH + kb) * 512) + loff;
          const _Float16* p2 = whht + (((size_t)(2 * JBLK + jb) * KBH + kb) * 512) + loff;
          v16h a0 = ldfrag(p0), a1 = ldfrag(p1), a2 = ldfrag(p2);
          v16h b = ldfrag(hcur + kb * 32 + half * 16);
          for (kb += NWAVE; kb < KBH; kb += NWAVE) {
            p0 += wstride; p1 += wstride; p2 += wstride;
            v16h n0 = ldfrag(p0), n1 = ldfrag(p1), n2 = ldfrag(p2);
            v16h nb = ldfrag(hcur + kb * 32 + half * 16);
            ar = wmma16(a0, b, ar); az = wmma16(a1, b, az); ag = wmma16(a2, b, ag);
            a0 = n0; a1 = n1; a2 = n2; b = nb;
          }
          ar = wmma16(a0, b, ar); az = wmma16(a1, b, az); ag = wmma16(a2, b, ag);
        }
        if ((lane & 15) == 0) {
#pragma unroll
          for (int r = 0; r < 8; ++r) {
            s_part[3][wav][half * 8 + r] = ar[r];
            s_part[4][wav][half * 8 + r] = az[r];
            s_part[5][wav][half * 8 + r] = ag[r];
          }
        }
      }
      __syncthreads();
      if (tid < 16) {
        const int m = tid, row = jb * 16 + m;
        float g6[6];
#pragma unroll
        for (int p = 0; p < 6; ++p) {
          float s = 0.f;
#pragma unroll
          for (int w = 0; w < NWAVE; ++w) s += s_part[p][w][m];
          g6[p] = s;
        }
        float ir = g6[0] + b_ih[row], iz = g6[1] + b_ih[HIDS + row], ig = g6[2] + b_ih[2 * HIDS + row];
        float hr = g6[3] + b_hh[row], hz = g6[4] + b_hh[HIDS + row], hg = g6[5] + b_hh[2 * HIDS + row];
        float rg = 1.f / (1.f + expf(-(ir + hr)));
        float zg = 1.f / (1.f + expf(-(iz + hz)));
        float ng = tanhf(ig + rg * hg);
        float hn = (1.f - zg) * ng + zg * hcur32[row];
        hnxt32[row] = hn;
        hnxt[row]   = (_Float16)hn;
      }
      __syncthreads();
    }
    gbar(cnt, rel, ++ep);

    // ---- Phase D: l2 = relu(W2 @ h_new + b2), WGs 0..47, sw-pipelined ----
    if (blockIdx.x < RB2) {
      const int rb = blockIdx.x;
      v8f acc = {};
      int kb = wav;
      if (kb < KB2) {
        const _Float16* p0 = w2t + (((size_t)rb * KB2 + kb) * 512) + loff;
        v16h a = ldfrag(p0);
        v16h b = ldfrag(hnxt + kb * 32 + half * 16);
        for (kb += NWAVE; kb < KB2; kb += NWAVE) {
          p0 += wstride;
          v16h na = ldfrag(p0);
          v16h nb = ldfrag(hnxt + kb * 32 + half * 16);
          acc = wmma16(a, b, acc);
          a = na; b = nb;
        }
        acc = wmma16(a, b, acc);
      }
      if ((lane & 15) == 0) {
#pragma unroll
        for (int r = 0; r < 8; ++r) s_part[0][wav][half * 8 + r] = acc[r];
      }
      __syncthreads();
      if (tid < 16) {
        int row = rb * 16 + tid;
        float s = 0.f;
#pragma unroll
        for (int w = 0; w < NWAVE; ++w) s += s_part[0][w][tid];
        s += b2[row];
        l2a[row] = (_Float16)(s > 0.f ? s : 0.f);
      }
      __syncthreads();
    }
    gbar(cnt, rel, ++ep);

    // ---- Phase E (WG0): Kalman gain, posterior, out[t], features for t+1 ----
    if (blockIdx.x == 0) {
      for (int rb = wav; rb < RB3; rb += NWAVE) {
        v8f acc = {};
        const _Float16* p0 = w3t + ((size_t)rb * KB3) * 512 + loff;
        v16h a = ldfrag(p0);
        v16h b = ldfrag(l2a + half * 16);
        for (int kb = 1; kb < KB3; ++kb) {
          p0 += 512;
          v16h na = ldfrag(p0);
          v16h nb = ldfrag(l2a + kb * 32 + half * 16);
          acc = wmma16(a, b, acc);
          a = na; b = nb;
        }
        acc = wmma16(a, b, acc);
        if ((lane & 15) == 0) {
#pragma unroll
          for (int r = 0; r < 8; ++r) {
            int row = rb * 16 + half * 8 + r;
            s_kg[row] = (acc[r] + b3[row]) * 1e-4f;
          }
        }
      }
      __syncthreads();
      if (tid < MM) {
        float s = 0.f;
        for (int n = 0; n < NNY; ++n) s += s_kg[tid * NNY + n] * s_resid[n];
        float xn = s_xprior[tid] + s;
        s_xpost[tid] = xn;
        dout[tid * TT + t] = xn;
      }
      __syncthreads();
      if (t + 1 < TT)
        phaseA(t + 1, yseq, Aa, Cc, s_xpost, s_xprior, s_resid, s_d4, s_inv, kgin);
    }
    gbar(cnt, rel, ++ep);
  }
}

extern "C" void kernel_launch(void* const* d_in, const int* in_sizes, int n_in,
                              void* d_out, int out_size, void* d_ws, size_t ws_size,
                              hipStream_t stream) {
  (void)in_sizes; (void)n_in; (void)out_size; (void)ws_size;
  const float* yseq = (const float*)d_in[0];
  const float* x0   = (const float*)d_in[1];
  const float* h0   = (const float*)d_in[2];
  const float* Aa   = (const float*)d_in[3];
  const float* Cc   = (const float*)d_in[4];
  const float* W1   = (const float*)d_in[5];
  const float* b1   = (const float*)d_in[6];
  const float* Wih  = (const float*)d_in[7];
  const float* Whh  = (const float*)d_in[8];
  const float* bih  = (const float*)d_in[9];
  const float* bhh  = (const float*)d_in[10];
  const float* W2   = (const float*)d_in[11];
  const float* b2   = (const float*)d_in[12];
  const float* W3   = (const float*)d_in[13];
  const float* b3   = (const float*)d_in[14];
  float* dout = (float*)d_out;

  char* ws = (char*)d_ws;
  size_t off = 0;
  auto take = [&](size_t bytes) {
    off = (off + 255) & ~(size_t)255;
    size_t o = off; off += bytes; return o;
  };
  size_t o_ctrl = take(256);
  size_t o_kgin = take(64 * sizeof(_Float16));
  size_t o_l1   = take((size_t)H1S * sizeof(_Float16));
  size_t o_l2   = take((size_t)H2S * sizeof(_Float16));
  size_t o_h16a = take((size_t)HPAD * sizeof(_Float16));
  size_t o_h16b = take((size_t)HPAD * sizeof(_Float16));
  size_t o_h32a = take((size_t)HIDS * sizeof(float));
  size_t o_h32b = take((size_t)HIDS * sizeof(float));
  size_t o_w1  = take((size_t)RB1 * KB1 * 512 * sizeof(_Float16));
  size_t o_wih = take((size_t)RBI * KBI * 512 * sizeof(_Float16));
  size_t o_whh = take((size_t)RBH * KBH * 512 * sizeof(_Float16));
  size_t o_w2  = take((size_t)RB2 * KB2 * 512 * sizeof(_Float16));
  size_t o_w3  = take((size_t)RB3 * KB3 * 512 * sizeof(_Float16));

  unsigned*  ctrl = (unsigned*)(ws + o_ctrl);
  _Float16*  kgin = (_Float16*)(ws + o_kgin);
  _Float16*  l1a  = (_Float16*)(ws + o_l1);
  _Float16*  l2a  = (_Float16*)(ws + o_l2);
  _Float16*  h16a = (_Float16*)(ws + o_h16a);
  _Float16*  h16b = (_Float16*)(ws + o_h16b);
  float*     h32a = (float*)(ws + o_h32a);
  float*     h32b = (float*)(ws + o_h32b);
  _Float16*  w1t  = (_Float16*)(ws + o_w1);
  _Float16*  wiht = (_Float16*)(ws + o_wih);
  _Float16*  whht = (_Float16*)(ws + o_whh);
  _Float16*  w2t  = (_Float16*)(ws + o_w2);
  _Float16*  w3t  = (_Float16*)(ws + o_w3);

  auto prep = [&](const float* W, int rows, int cols, int RB, int KB, _Float16* dst) {
    long threads = (long)RB * KB * 32;
    unsigned grid = (unsigned)((threads + 255) / 256);
    knet_prep<<<grid, 256, 0, stream>>>(W, rows, cols, RB, KB, dst);
  };
  prep(W1,  H1S,      MM + NNY, RB1, KB1, w1t);
  prep(Wih, 3 * HIDS, H1S,      RBI, KBI, wiht);
  prep(Whh, 3 * HIDS, HIDS,     RBH, KBH, whht);
  prep(W2,  H2S,      HIDS,     RB2, KB2, w2t);
  prep(W3,  OUTR,     H2S,      RB3, KB3, w3t);

  knet_init<<<(HPAD + 255) / 256, 256, 0, stream>>>(h0, h32a, h16a, h16b, ctrl);

  knet_main<<<GWG, 256, 0, stream>>>(yseq, x0, Aa, Cc, b1, bih, bhh, b2, b3,
                                     ctrl, kgin, l1a, l2a, h16a, h16b, h32a, h32b,
                                     w1t, wiht, whht, w2t, w3t, dout);
}
